// Attention_81071802679592
// MI455X (gfx1250) — compile-verified, matrix operand order checked by reading, not run on hardware
//
#include <hip/hip_runtime.h>

#define B_ 4
#define L_ 2048
#define C_ 1024
#define H_ 16
#define D_ 64

typedef unsigned short u16;
typedef unsigned int   u32;
typedef unsigned long long u64;

typedef __attribute__((ext_vector_type(16))) __bf16 bf16x16;
typedef __attribute__((ext_vector_type(8)))  float  floatx8;
typedef __attribute__((ext_vector_type(4)))  unsigned int u32x4;
typedef __attribute__((ext_vector_type(8)))  int i32x8;
typedef __attribute__((ext_vector_type(4)))  int i32x4;

union FragAB {
  bf16x16 v;
  uint4   q[2];
  u16     u[16];
};
union FragC {
  floatx8 v;
  float   f[8];
};
union U2 {
  uint2 d;
  u16   u[4];
};

__device__ __forceinline__ u16 f2bf(float x) {
  u32 u = __float_as_uint(x);
  u32 r = u + 0x7FFFu + ((u >> 16) & 1u);   // round-to-nearest-even
  return (u16)(r >> 16);
}
__device__ __forceinline__ float bf2f(u16 h) {
  return __uint_as_float(((u32)h) << 16);
}

__device__ __forceinline__ bf16x16 load16(const u16* p) {
  FragAB f;
  f.q[0] = *reinterpret_cast<const uint4*>(p);
  f.q[1] = *reinterpret_cast<const uint4*>(p + 8);
  return f.v;
}

__device__ __forceinline__ floatx8 wmma_bf16(bf16x16 a, bf16x16 b, floatx8 c) {
  return __builtin_amdgcn_wmma_f32_16x16x32_bf16(false, a, false, b, (short)0, c,
                                                 false, false);
}

// ---------------------------------------------------------------- convert
__global__ __launch_bounds__(256) void cvt_f32_bf16(const float* __restrict__ src,
                                                    u16* __restrict__ dst, int n4) {
  int i = blockIdx.x * 256 + threadIdx.x;
  if (i >= n4) return;
  float4 f = reinterpret_cast<const float4*>(src)[i];
  U2 o;
  o.u[0] = f2bf(f.x); o.u[1] = f2bf(f.y); o.u[2] = f2bf(f.z); o.u[3] = f2bf(f.w);
  reinterpret_cast<uint2*>(dst)[i] = o.d;
}

// ---------------------------------------------------------------- QKV GEMM
// qkv[m,n] = sum_k X[m,k] * Wqkv[n,k] + bias(n)
// q,k -> [B,H,L,D] bf16 ; v -> [B,H,D,L] bf16 (pre-transposed for attention B-frags)
__device__ __forceinline__ void store_qkv_tile(const FragC& c, int r0, int n,
                                               const float* __restrict__ qbias,
                                               const float* __restrict__ vbias,
                                               u16* __restrict__ qb,
                                               u16* __restrict__ kb,
                                               u16* __restrict__ vtb) {
  int part = n >> 10;
  int cc = n & (C_ - 1);
  int h = cc >> 6, d = cc & 63;
  float bias = (part == 0) ? qbias[cc] : ((part == 2) ? vbias[cc] : 0.0f);
#pragma unroll
  for (int i = 0; i < 8; ++i) {
    int m = r0 + i;
    int b = m >> 11, l = m & (L_ - 1);
    u16 val = f2bf(c.f[i] + bias);
    if (part == 2) {
      vtb[(((size_t)(b * H_ + h)) * D_ + d) * L_ + l] = val;
    } else {
      u16* dst = (part == 0) ? qb : kb;
      dst[(((size_t)(b * H_ + h)) * L_ + l) * D_ + d] = val;
    }
  }
}

__global__ __launch_bounds__(256) void gemm_qkv(const u16* __restrict__ X,
                                                const u16* __restrict__ W,
                                                const float* __restrict__ qbias,
                                                const float* __restrict__ vbias,
                                                u16* __restrict__ qb,
                                                u16* __restrict__ kb,
                                                u16* __restrict__ vtb) {
  const int K = C_;
  int wave = threadIdx.x >> 5, lane = threadIdx.x & 31;
  int wm = wave & 1, wn = wave >> 1;          // 2 (M) x 4 (N) waves
  int m0 = blockIdx.y * 128 + wm * 64;        // 64-row wave tile
  int n0 = blockIdx.x * 128 + wn * 32;        // 32-col wave tile
  int lm = lane & 15, lk = (lane >> 4) * 16;

  const u16* A0 = X + (size_t)(m0 + lm) * K + lk;
  const u16* B0 = W + (size_t)(n0 + lm) * K + lk;

  floatx8 z = {};
  FragC c[4][2];
#pragma unroll
  for (int r = 0; r < 4; ++r) { c[r][0].v = z; c[r][1].v = z; }

  for (int k = 0; k < K; k += 32) {
    bf16x16 a[4];
#pragma unroll
    for (int r = 0; r < 4; ++r) a[r] = load16(A0 + (size_t)(r * 16) * K + k);
    bf16x16 b0 = load16(B0 + k);
    bf16x16 b1 = load16(B0 + (size_t)16 * K + k);
    __builtin_prefetch(A0 + k + 128, 0, 3);
    __builtin_prefetch(B0 + k + 128, 0, 3);
#pragma unroll
    for (int r = 0; r < 4; ++r) {
      c[r][0].v = wmma_bf16(a[r], b0, c[r][0].v);
      c[r][1].v = wmma_bf16(a[r], b1, c[r][1].v);
    }
  }

  int colA = n0 + lm;
#pragma unroll
  for (int r = 0; r < 4; ++r) {
    int r0 = m0 + r * 16 + (lane >> 4) * 8;
    store_qkv_tile(c[r][0], r0, colA,      qbias, vbias, qb, kb, vtb);
    store_qkv_tile(c[r][1], r0, colA + 16, qbias, vbias, qb, kb, vtb);
  }
}

// ---------------------------------------------------------------- q/k row L2 norm
__global__ __launch_bounds__(256) void norm_qk(u16* __restrict__ qb,
                                               u16* __restrict__ kb,
                                               const float* __restrict__ sml) {
  const size_t NR = (size_t)B_ * H_ * L_;
  int wave = threadIdx.x >> 5, lane = threadIdx.x & 31;
  size_t gid = (size_t)blockIdx.x * 8 + wave;
  bool isQ = gid < NR;
  size_t row = isQ ? gid : gid - NR;
  u16* base = (isQ ? qb : kb) + row * D_ + lane * 2;
  float x0 = bf2f(base[0]);
  float x1 = bf2f(base[1]);
  float ss = x0 * x0 + x1 * x1;
#pragma unroll
  for (int m = 1; m < 32; m <<= 1) ss += __shfl_xor(ss, m, 32);
  float scale = 1.0f / fmaxf(sqrtf(ss), 1e-12f);
  if (isQ) {
    int h = (int)((row >> 11) & (H_ - 1));
    scale *= __expf(fminf(sml[h], 4.605170185988091f));  // log(100)
  }
  base[0] = f2bf(x0 * scale);
  base[1] = f2bf(x1 * scale);
}

// ---------------------------------------------------------------- flash attention
// K tile: TDM tensor_load_to_lds (wave 0, double buffered, TENSORcnt)
// Vt tile: per-lane global_load_async_to_lds_b128 (ASYNCcnt)
__global__ __launch_bounds__(128) void attn_fa(const u16* __restrict__ qb,
                                               const u16* __restrict__ kb,
                                               const u16* __restrict__ vtb,
                                               u16* __restrict__ ob) {
  __shared__ u16 sK[2][64][72];    // K tile   [kv][d], 144B padded rows
  __shared__ u16 sVt[2][64][72];   // V^T tile [d][kv]
  __shared__ u16 sP[4][16][72];    // per-wave probs [m][kv]

  const int NT = L_ / 64;
  int bh = blockIdx.y;
  int qt = blockIdx.x;
  int wave = threadIdx.x >> 5, lane = threadIdx.x & 31;
  int b = bh >> 4, h = bh & (H_ - 1);
  int lm = lane & 15, lk = (lane >> 4) * 16;

  const u16* Q  = qb  + (size_t)bh * L_ * D_;
  const u16* Kp = kb  + (size_t)bh * L_ * D_;
  const u16* Vt = vtb + (size_t)bh * D_ * L_;

  // TDM descriptor load of K tile t into buffer buf (issued by wave 0 only).
  auto issue_k_tdm = [&](int t, int buf) {
    u64 ga = (u64)(uintptr_t)(Kp + (size_t)t * 64 * D_);
    u32x4 g0;
    g0[0] = 1u;                                   // count = 1 valid descriptor
    g0[1] = (u32)(uintptr_t)&sK[buf][0][0];       // lds_addr
    g0[2] = (u32)ga;                              // global_addr lo
    g0[3] = (u32)(ga >> 32) | 0x80000000u;        // global_addr hi | type=2
    i32x8 g1;
    g1[0] = (int)(0x00010000u |                   // data_size = 2 bytes
                  (1u << 20) |                    // pad_enable
                  (4u << 22) |                    // pad_interval: 32 DWORDs (=128B row)
                  (3u << 25));                    // pad_amount: 4 DWORDs (=16B -> 144B stride)
    g1[1] = (int)(64u << 16);                     // tensor_dim0 = 64 (bits 79:48)
    g1[2] = (int)(2048u << 16);                   // tensor_dim1 = 2048 (bits 111:80)
    g1[3] = (int)(64u << 16);                     // tile_dim0 = 64 (bits 127:112)
    g1[4] = 64;                                   // tile_dim1 = 64 (bits 143:128)
    g1[5] = 64;                                   // tensor_dim0_stride = 64 (bits 207:160)
    g1[6] = 0;
    g1[7] = 0;
    i32x4 gz4 = {};
    i32x8 gz8 = {};
    __builtin_amdgcn_tensor_load_to_lds(g0, g1, gz4, gz4, gz8, 0);
  };

  // Per-lane async global->LDS fill of V^T tile t into buffer buf (4 x b128 per lane).
  auto issue_vt = [&](int t, int buf) {
    u32 lbase = (u32)(uintptr_t)&sVt[buf][0][0];
#pragma unroll
    for (int cch = 0; cch < 4; ++cch) {
      int id = (int)threadIdx.x + cch * 128;      // 0..511
      int r = id >> 3, cq = id & 7;               // d-row, 8-elem chunk
      const u16* g = Vt + (size_t)r * L_ + t * 64 + cq * 8;
      u32 laddr = lbase + (u32)(r * 144 + cq * 16);
      asm volatile("global_load_async_to_lds_b128 %0, %1, off"
                   :: "v"(laddr), "v"(g) : "memory");
    }
  };

  int qrow0 = qt * 64 + wave * 16;
  bf16x16 aq0 = load16(Q + (size_t)(qrow0 + lm) * D_ + lk);
  bf16x16 aq1 = load16(Q + (size_t)(qrow0 + lm) * D_ + lk + 32);

  floatx8 z = {};
  FragC o[4];
#pragma unroll
  for (int j = 0; j < 4; ++j) o[j].v = z;
  float mrow[8], lrow[8];
#pragma unroll
  for (int i = 0; i < 8; ++i) { mrow[i] = -1e30f; lrow[i] = 0.0f; }

  // prologue: start DMA of tile 0
  if (wave == 0) issue_k_tdm(0, 0);
  issue_vt(0, 0);

  for (int t = 0; t < NT; ++t) {
    int buf = t & 1;
    bool hn = (t + 1) < NT;
    if (hn) {
      if (wave == 0) issue_k_tdm(t + 1, buf ^ 1);
      issue_vt(t + 1, buf ^ 1);
    }
    if (wave == 0) {
      if (hn) __builtin_amdgcn_s_wait_tensorcnt(1);
      else    __builtin_amdgcn_s_wait_tensorcnt(0);
    }
    if (hn) asm volatile("s_wait_asynccnt 0x4" ::: "memory");
    else    asm volatile("s_wait_asynccnt 0x0" ::: "memory");
    __syncthreads();   // tile t visible to all waves

    // S = q_hat @ k_hat^T  (16 x 64), scale = 1
    FragC s[4];
#pragma unroll
    for (int j = 0; j < 4; ++j) {
      bf16x16 bk0 = load16(&sK[buf][j * 16 + lm][lk]);
      bf16x16 bk1 = load16(&sK[buf][j * 16 + lm][lk + 32]);
      floatx8 acc = z;
      acc = wmma_bf16(aq0, bk0, acc);
      acc = wmma_bf16(aq1, bk1, acc);
      s[j].v = acc;
    }

    // online softmax (row stats per vgpr-index i within 16-lane halves)
#pragma unroll
    for (int i = 0; i < 8; ++i) {
      float pm = fmaxf(fmaxf(s[0].f[i], s[1].f[i]), fmaxf(s[2].f[i], s[3].f[i]));
#pragma unroll
      for (int m = 1; m < 16; m <<= 1) pm = fmaxf(pm, __shfl_xor(pm, m, 32));
      float mn = fmaxf(mrow[i], pm);
      float sc = __expf(mrow[i] - mn);
      mrow[i] = mn;
#pragma unroll
      for (int j = 0; j < 4; ++j) o[j].f[i] *= sc;
      float rs = 0.0f;
#pragma unroll
      for (int j = 0; j < 4; ++j) {
        float p = __expf(s[j].f[i] - mn);
        s[j].f[i] = p;
        rs += p;
      }
#pragma unroll
      for (int m = 1; m < 16; m <<= 1) rs += __shfl_xor(rs, m, 32);
      lrow[i] = lrow[i] * sc + rs;
    }

    // stage P into LDS to re-stripe into A-fragment layout
#pragma unroll
    for (int j = 0; j < 4; ++j)
#pragma unroll
      for (int i = 0; i < 8; ++i)
        sP[wave][(lane >> 4) * 8 + i][j * 16 + lm] = f2bf(s[j].f[i]);

    // O += P @ V  (V^T tile already in B-frag friendly [d][kv] layout)
#pragma unroll
    for (int kk = 0; kk < 2; ++kk) {
      bf16x16 ap = load16(&sP[wave][lm][lk + kk * 32]);
#pragma unroll
      for (int j = 0; j < 4; ++j) {
        bf16x16 bv = load16(&sVt[buf][j * 16 + lm][lk + kk * 32]);
        o[j].v = wmma_bf16(ap, bv, o[j].v);
      }
    }
    __syncthreads();   // everyone done with buf before it is refilled at t+2
  }

  // finalize: divide by row sum, write [B, L, C] bf16
#pragma unroll
  for (int i = 0; i < 8; ++i) {
    int r = qrow0 + (lane >> 4) * 8 + i;
    float inv = 1.0f / fmaxf(lrow[i], 1e-30f);
#pragma unroll
    for (int j = 0; j < 4; ++j) {
      int d = j * 16 + lm;
      ob[((size_t)b * L_ + r) * C_ + h * D_ + d] = f2bf(o[j].f[i] * inv);
    }
  }
}

// ---------------------------------------------------------------- output projection
__global__ __launch_bounds__(256) void gemm_proj(const u16* __restrict__ A,
                                                 const u16* __restrict__ W,
                                                 const float* __restrict__ bias,
                                                 float* __restrict__ out) {
  const int K = C_;
  int wave = threadIdx.x >> 5, lane = threadIdx.x & 31;
  int wm = wave & 1, wn = wave >> 1;
  int m0 = blockIdx.y * 128 + wm * 64;
  int n0 = blockIdx.x * 128 + wn * 32;
  int lm = lane & 15, lk = (lane >> 4) * 16;

  const u16* A0 = A + (size_t)(m0 + lm) * K + lk;
  const u16* B0 = W + (size_t)(n0 + lm) * K + lk;

  floatx8 z = {};
  FragC c[4][2];
#pragma unroll
  for (int r = 0; r < 4; ++r) { c[r][0].v = z; c[r][1].v = z; }

  for (int k = 0; k < K; k += 32) {
    bf16x16 a[4];
#pragma unroll
    for (int r = 0; r < 4; ++r) a[r] = load16(A0 + (size_t)(r * 16) * K + k);
    bf16x16 b0 = load16(B0 + k);
    bf16x16 b1 = load16(B0 + (size_t)16 * K + k);
    __builtin_prefetch(A0 + k + 128, 0, 3);
    __builtin_prefetch(B0 + k + 128, 0, 3);
#pragma unroll
    for (int r = 0; r < 4; ++r) {
      c[r][0].v = wmma_bf16(a[r], b0, c[r][0].v);
      c[r][1].v = wmma_bf16(a[r], b1, c[r][1].v);
    }
  }

  int colA = n0 + lm;
  int colB = colA + 16;
  float bbA = bias[colA], bbB = bias[colB];
#pragma unroll
  for (int r = 0; r < 4; ++r) {
    int r0 = m0 + r * 16 + (lane >> 4) * 8;
#pragma unroll
    for (int i = 0; i < 8; ++i) {
      out[(size_t)(r0 + i) * C_ + colA] = c[r][0].f[i] + bbA;
      out[(size_t)(r0 + i) * C_ + colB] = c[r][1].f[i] + bbB;
    }
  }
}

// ---------------------------------------------------------------- launch
extern "C" void kernel_launch(void* const* d_in, const int* in_sizes, int n_in,
                              void* d_out, int out_size, void* d_ws, size_t ws_size,
                              hipStream_t stream) {
  (void)in_sizes; (void)n_in; (void)out_size; (void)ws_size;

  const float* x     = (const float*)d_in[0];
  const float* Wqkv  = (const float*)d_in[1];
  const float* qbias = (const float*)d_in[2];
  const float* vbias = (const float*)d_in[3];
  const float* sml   = (const float*)d_in[4];
  const float* Wproj = (const float*)d_in[5];
  const float* bproj = (const float*)d_in[6];
  float* out = (float*)d_out;

  char* ws = (char*)d_ws;
  size_t off = 0;
  auto alloc = [&](size_t bytes) -> void* {
    void* p = ws + off;
    off += (bytes + 255) & ~(size_t)255;
    return p;
  };
  u16* xb     = (u16*)alloc((size_t)B_ * L_ * C_ * 2);
  u16* wqkvb  = (u16*)alloc((size_t)3 * C_ * C_ * 2);
  u16* wprojb = (u16*)alloc((size_t)C_ * C_ * 2);
  u16* qb     = (u16*)alloc((size_t)B_ * H_ * L_ * D_ * 2);
  u16* kb     = (u16*)alloc((size_t)B_ * H_ * L_ * D_ * 2);
  u16* vtb    = (u16*)alloc((size_t)B_ * H_ * D_ * L_ * 2);
  u16* attnb  = (u16*)alloc((size_t)B_ * L_ * C_ * 2);

  int nx4 = B_ * L_ * C_ / 4;
  cvt_f32_bf16<<<(nx4 + 255) / 256, 256, 0, stream>>>(x, xb, nx4);
  int nw4 = 3 * C_ * C_ / 4;
  cvt_f32_bf16<<<(nw4 + 255) / 256, 256, 0, stream>>>(Wqkv, wqkvb, nw4);
  int np4 = C_ * C_ / 4;
  cvt_f32_bf16<<<(np4 + 255) / 256, 256, 0, stream>>>(Wproj, wprojb, np4);

  dim3 g1(3 * C_ / 128, B_ * L_ / 128);
  gemm_qkv<<<g1, 256, 0, stream>>>(xb, wqkvb, qbias, vbias, qb, kb, vtb);

  int nrows = 2 * B_ * H_ * L_;
  norm_qk<<<nrows / 8, 256, 0, stream>>>(qb, kb, sml);

  dim3 g2(L_ / 64, B_ * H_);
  attn_fa<<<g2, 128, 0, stream>>>(qb, kb, vtb, attnb);

  dim3 g3(C_ / 128, B_ * L_ / 128);
  gemm_proj<<<g3, 256, 0, stream>>>(attnb, wprojb, bproj, out);
}
